// ComplexTGNLayer_30880814858911
// MI455X (gfx1250) — matrix operand order repaired and checked by reading, not compiled, not run on hardware
//
#include <hip/hip_runtime.h>
#include <stdint.h>

// ---------------------------------------------------------------------------
// ComplexTGN layer for MI455X (gfx1250, wave32, WMMA bf16 16x16x32).
//
// All complex linears are lowered to ONE real bf16 GEMM each via the packing
//   A    = [Xr | Xi]                      (M x 2K)   bf16
//   Wbig = [[Wr, -Wi], [Wi, Wr]]         (2O x 2K)  bf16   (packed per call)
//   ybig = A * Wbig^T + [br; bi]         (M x 2O)   f32 accum
// GEMM tiles are staged with GLOBAL_LOAD_ASYNC_TO_LDS_B128 (ASYNCcnt) into a
// double-buffered LDS, overlapping next-tile DMA with current-tile WMMA.
// Attention is flash-style with WMMA for Q·K^H and P·V.
// ---------------------------------------------------------------------------

#define Bsz   4
#define Tseq  1024
#define Dm    1024
#define Hh    16
#define HDd   64
#define DFFd  4096
#define Mtok  (Bsz * Tseq)   // 4096 tokens

typedef __attribute__((ext_vector_type(16))) __bf16 v16bf;
typedef __attribute__((ext_vector_type(8)))  float  v8f;

union Frag { uint4 q[2]; v16bf v; unsigned short s[16]; };

__device__ inline unsigned short f2b(float f) {
  unsigned int u = __builtin_bit_cast(unsigned int, f);
  u += 0x7FFFu + ((u >> 16) & 1u);      // round-to-nearest-even
  return (unsigned short)(u >> 16);
}

__device__ inline v8f v8zero() {
  v8f z;
#pragma unroll
  for (int i = 0; i < 8; i++) z[i] = 0.f;
  return z;
}

__device__ inline v8f wmma_bf16(v16bf a, v16bf b, v8f c) {
  // emits v_wmma_f32_16x16x32_bf16
  return __builtin_amdgcn_wmma_f32_16x16x32_bf16(false, a, false, b,
                                                 (short)0, c, false, false);
}

// --- CDNA5 async memory->LDS DMA (CDNA5 ISA, ASYNCcnt) -----
// Per lane: 16B from global addr into LDS at byte offset held in a VGPR.
__device__ inline void async_b128(unsigned lds_off, const void* gaddr) {
  asm volatile("global_load_async_to_lds_b128 %0, %1, off"
               :: "v"(lds_off), "v"(gaddr) : "memory");
}
__device__ inline void wait_async_le4() {
  asm volatile("s_wait_asynccnt 4" ::: "memory");
}
__device__ inline void wait_async_0() {
  asm volatile("s_wait_asynccnt 0" ::: "memory");
}
__device__ inline unsigned lds_off_of(const void* p) {
  // flat shared pointer: LDS offset lives in the low 32 bits (aperture rule)
  return (unsigned)(unsigned long long)p;
}

// ---------------------------------------------------------------------------
// Weight packing: W[2][O][K] f32 -> Wbig[2O][2K] bf16
// ---------------------------------------------------------------------------
__global__ void pack_w_kernel(const float* __restrict__ W,
                              unsigned short* __restrict__ Wb, int O, int K) {
  size_t idx = (size_t)blockIdx.x * 256 + threadIdx.x;
  size_t total = 4ull * O * K;
  if (idx >= total) return;
  int K2 = 2 * K;
  int r = (int)(idx / K2), c = (int)(idx % K2);
  int qa = (r >= O) ? 1 : 0, qb = (c >= K) ? 1 : 0;
  int rr = r - qa * O, cc = c - qb * K;
  size_t plane = (size_t)O * K;
  float val;
  if (qa == qb)      val =  W[(size_t)rr * K + cc];          // Wr (diag blocks)
  else if (qb)       val = -W[plane + (size_t)rr * K + cc];  // top-right: -Wi
  else               val =  W[plane + (size_t)rr * K + cc];  // bot-left:   Wi
  Wb[idx] = f2b(val);
}

// ---------------------------------------------------------------------------
// Complex LayerNorm (per real/imag component) -> packed bf16 A = [Xr|Xi]
// ---------------------------------------------------------------------------
__global__ __launch_bounds__(256) void cln_kernel(
    const float* __restrict__ zr, const float* __restrict__ zi,
    const float* __restrict__ gam, const float* __restrict__ bet,
    unsigned short* __restrict__ Aout) {
  __shared__ float red0[256], red1[256];
  int m = blockIdx.x, tid = threadIdx.x;
  float xr[4], xi[4];
  float s0 = 0.f, s1 = 0.f;
#pragma unroll
  for (int j = 0; j < 4; j++) {
    int d = tid + j * 256;
    xr[j] = zr[(size_t)m * Dm + d];
    xi[j] = zi[(size_t)m * Dm + d];
    s0 += xr[j]; s1 += xi[j];
  }
  red0[tid] = s0; red1[tid] = s1; __syncthreads();
  for (int s = 128; s > 0; s >>= 1) {
    if (tid < s) { red0[tid] += red0[tid + s]; red1[tid] += red1[tid + s]; }
    __syncthreads();
  }
  float mr = red0[0] * (1.f / Dm), mi = red1[0] * (1.f / Dm);
  __syncthreads();
  s0 = 0.f; s1 = 0.f;
#pragma unroll
  for (int j = 0; j < 4; j++) {
    float a = xr[j] - mr, b = xi[j] - mi;
    s0 += a * a; s1 += b * b;
  }
  red0[tid] = s0; red1[tid] = s1; __syncthreads();
  for (int s = 128; s > 0; s >>= 1) {
    if (tid < s) { red0[tid] += red0[tid + s]; red1[tid] += red1[tid + s]; }
    __syncthreads();
  }
  float rr = rsqrtf(red0[0] * (1.f / Dm) + 1e-5f);
  float ri = rsqrtf(red1[0] * (1.f / Dm) + 1e-5f);
#pragma unroll
  for (int j = 0; j < 4; j++) {
    int d = tid + j * 256;
    float yr = (xr[j] - mr) * rr * gam[d] + bet[d];
    float yi = (xi[j] - mi) * ri * gam[Dm + d] + bet[Dm + d];
    Aout[(size_t)m * (2 * Dm) + d]      = f2b(yr);
    Aout[(size_t)m * (2 * Dm) + Dm + d] = f2b(yi);
  }
}

// ---------------------------------------------------------------------------
// Generic bf16 WMMA GEMM: C[M,N] = A[M,K] * W[N,K]^T + bias[N]
// Block 128x128, BK=32, 256 threads = 8 waves in 4(M) x 2(N); each wave does
// a 32x64 tile = 2x4 v_wmma_f32_16x16x32_bf16 per k-step.
// Tiles staged by async DMA into double-buffered LDS:
//   issue(k+1) -> s_wait_asynccnt 4 (tile k resident) -> barrier -> WMMA(k)
// mode 0: f32 out; mode 1: relu->bf16 out; mode 2: f32 + bf16 out.
// ---------------------------------------------------------------------------
#define GBM 128
#define GBN 128
#define GBK 32
#define ASTR 40   // bf16 LDS row stride (pad to dodge bank conflicts)

__global__ __launch_bounds__(256) void gemm_cplx_bf16(
    const unsigned short* __restrict__ A, const unsigned short* __restrict__ W,
    const float* __restrict__ bias, float* __restrict__ Cf,
    unsigned short* __restrict__ Cb, int Nd, int Kd, int mode) {
  __shared__ __align__(16) unsigned short As[2][GBM * ASTR];
  __shared__ __align__(16) unsigned short Bs[2][GBN * ASTR];
  const int tid = threadIdx.x;
  const int lane = tid & 31, wv = tid >> 5;
  const int wm = wv & 3, wn = wv >> 2;
  const int lm = lane & 15, lh = lane >> 4;
  const int bm0 = blockIdx.y * GBM, bn0 = blockIdx.x * GBN;

  v8f acc[2][4];
#pragma unroll
  for (int i = 0; i < 2; i++)
#pragma unroll
    for (int j = 0; j < 4; j++) acc[i][j] = v8zero();

  // each thread DMAs one 32B slice of the A tile and one of the B tile
  const int srow = tid >> 1, shalf = tid & 1;
  const unsigned short* gaBase = A + (size_t)(bm0 + srow) * Kd + shalf * 16;
  const unsigned short* gbBase = W + (size_t)(bn0 + srow) * Kd + shalf * 16;
  const unsigned ldsA[2] = { lds_off_of(&As[0][srow * ASTR + shalf * 16]),
                             lds_off_of(&As[1][srow * ASTR + shalf * 16]) };
  const unsigned ldsB[2] = { lds_off_of(&Bs[0][srow * ASTR + shalf * 16]),
                             lds_off_of(&Bs[1][srow * ASTR + shalf * 16]) };

  // prologue: DMA tile 0 into buffer 0 (4 async ops per wave)
  async_b128(ldsA[0],      gaBase);
  async_b128(ldsA[0] + 16, gaBase + 8);
  async_b128(ldsB[0],      gbBase);
  async_b128(ldsB[0] + 16, gbBase + 8);

  int buf = 0;
  for (int k0 = 0; k0 < Kd; k0 += GBK) {
    if (k0 + GBK < Kd) {               // DMA next tile into the other buffer
      int nb = buf ^ 1;
      async_b128(ldsA[nb],      gaBase + k0 + GBK);
      async_b128(ldsA[nb] + 16, gaBase + k0 + GBK + 8);
      async_b128(ldsB[nb],      gbBase + k0 + GBK);
      async_b128(ldsB[nb] + 16, gbBase + k0 + GBK + 8);
      wait_async_le4();                // the 4 older ops (tile k0) are done
    } else {
      wait_async_0();
    }
    __syncthreads();                   // tile k0 visible to all waves

    Frag af[2], bfr[4];
#pragma unroll
    for (int mt = 0; mt < 2; mt++) {
      // A frag (16x32): lanes 0-15 hold K 0-7 / 16-23, lanes 16-31 K 8-15 / 24-31
      const unsigned short* p = &As[buf][(wm * 32 + mt * 16 + lm) * ASTR + lh * 8];
      af[mt].q[0] = *(const uint4*)p;
      af[mt].q[1] = *(const uint4*)(p + 16);
    }
#pragma unroll
    for (int nt = 0; nt < 4; nt++) {
      // B frag (32x16): lanes 0-15 hold K 0-7 / 16-23, lanes 16-31 K 8-15 / 24-31
      const unsigned short* p = &Bs[buf][(wn * 64 + nt * 16 + lm) * ASTR + lh * 8];
      bfr[nt].q[0] = *(const uint4*)p;
      bfr[nt].q[1] = *(const uint4*)(p + 16);
    }
#pragma unroll
    for (int mt = 0; mt < 2; mt++)
#pragma unroll
      for (int nt = 0; nt < 4; nt++)
        acc[mt][nt] = wmma_bf16(af[mt].v, bfr[nt].v, acc[mt][nt]);

    __syncthreads();                   // all waves done reading buf
    buf ^= 1;
  }

#pragma unroll
  for (int mt = 0; mt < 2; mt++) {
#pragma unroll
    for (int nt = 0; nt < 4; nt++) {
      int gn = bn0 + wn * 64 + nt * 16 + lm;
      float bv = bias[gn];
#pragma unroll
      for (int v = 0; v < 8; v++) {
        int gm = bm0 + wm * 32 + mt * 16 + v + 8 * lh;
        float val = acc[mt][nt][v] + bv;
        size_t off = (size_t)gm * Nd + gn;
        if (mode == 0)      Cf[off] = val;
        else if (mode == 1) Cb[off] = f2b(fmaxf(val, 0.f));     // fused CReLU
        else { Cf[off] = val; Cb[off] = f2b(val); }
      }
    }
  }
}

// ---------------------------------------------------------------------------
// RoPE on q,k (+1/sqrt(HD) folded into q) and bf16 conversion of q,k,v.
// ---------------------------------------------------------------------------
__global__ void rope_convert_kernel(const float* __restrict__ Cq,
                                    const float* __restrict__ Ck,
                                    const float* __restrict__ Cv,
                                    unsigned short* __restrict__ qb,
                                    unsigned short* __restrict__ kb,
                                    unsigned short* __restrict__ vb) {
  int idx = blockIdx.x * 256 + threadIdx.x;  // over Mtok*Dm
  int m = idx >> 10, d = idx & 1023;
  int t = m & (Tseq - 1);
  int hd = d & (HDd - 1);
  float ang = (float)t * __powf(10000.f, -(float)hd * (1.f / HDd));
  float sn, cs;
  __sincosf(ang, &sn, &cs);
  size_t o_r = (size_t)m * (2 * Dm) + d, o_i = o_r + Dm;
  float qr = Cq[o_r], qi = Cq[o_i];
  qb[o_r] = f2b((qr * cs - qi * sn) * 0.125f);   // 1/sqrt(64)
  qb[o_i] = f2b((qr * sn + qi * cs) * 0.125f);
  float kr = Ck[o_r], ki = Ck[o_i];
  kb[o_r] = f2b(kr * cs - ki * sn);
  kb[o_i] = f2b(kr * sn + ki * cs);
  vb[o_r] = f2b(Cv[o_r]);
  vb[o_i] = f2b(Cv[o_i]);
}

// ---------------------------------------------------------------------------
// Flash attention, complex scores, softmax over |score| with causal mask.
// Grid (T/128, H, B); 256 threads = 8 waves; wave w owns queries [w*16,w*16+16).
// Key tile = 32; per tile: 16 WMMAs for scores + 8 WMMAs for P·V.
// ---------------------------------------------------------------------------
__global__ __launch_bounds__(256) void attn_kernel(
    const unsigned short* __restrict__ qb, const unsigned short* __restrict__ kb,
    const unsigned short* __restrict__ vb, unsigned short* __restrict__ aattn) {
  __shared__ __align__(16) unsigned short Ks[2][32 * 64];   // [r/i][key][hd]
  __shared__ __align__(16) unsigned short Vt[2][64 * 32];   // [r/i][hd][key]
  __shared__ __align__(16) unsigned short Ps[8][16 * 32];   // per-wave P tile
  const int b = blockIdx.z, h = blockIdx.y, qt = blockIdx.x;
  const int tid = threadIdx.x, lane = tid & 31, w = tid >> 5;
  const int lm = lane & 15, lh = lane >> 4;
  const int q0 = qt * 128 + w * 16;

  // Load Q fragments (Qr, Qi, -Qr), kept in registers for the whole loop.
  Frag qr[2], qi[2], qrn[2];
  {
    size_t rowoff = (size_t)(b * Tseq + q0 + lm) * (2 * Dm) + h * HDd;
#pragma unroll
    for (int c = 0; c < 2; c++) {
      const unsigned short* pr = qb + rowoff + c * 32;
      qr[c].q[0] = *(const uint4*)(pr + lh * 8);
      qr[c].q[1] = *(const uint4*)(pr + 16 + lh * 8);
      const unsigned short* pi = qb + rowoff + Dm + c * 32;
      qi[c].q[0] = *(const uint4*)(pi + lh * 8);
      qi[c].q[1] = *(const uint4*)(pi + 16 + lh * 8);
#pragma unroll
      for (int j = 0; j < 2; j++) {
        uint4 t = qr[c].q[j];
        t.x ^= 0x80008000u; t.y ^= 0x80008000u;
        t.z ^= 0x80008000u; t.w ^= 0x80008000u;  // bf16 sign flip
        qrn[c].q[j] = t;
      }
    }
  }

  v8f oacc[2][4];
#pragma unroll
  for (int c = 0; c < 2; c++)
#pragma unroll
    for (int n = 0; n < 4; n++) oacc[c][n] = v8zero();
  float mrun[8], lrun[8];
#pragma unroll
  for (int v = 0; v < 8; v++) { mrun[v] = -1e30f; lrun[v] = 0.f; }

  const int nkt = qt * 4 + 4;  // causal: keys <= qt*128+127
  for (int kt = 0; kt < nkt; kt++) {
    { // cooperative staging: K row-major, V transposed
      int comp = tid >> 7, r = tid & 127, key = r >> 2, hc = (r & 3) * 16;
      size_t src = (size_t)(b * Tseq + kt * 32 + key) * (2 * Dm)
                 + comp * Dm + h * HDd + hc;
      uint4 a0 = *(const uint4*)(kb + src);
      uint4 a1 = *(const uint4*)(kb + src + 8);
      *(uint4*)&Ks[comp][key * 64 + hc]     = a0;
      *(uint4*)&Ks[comp][key * 64 + hc + 8] = a1;
      uint4 v0 = *(const uint4*)(vb + src);
      uint4 v1 = *(const uint4*)(vb + src + 8);
      unsigned short buf[16];
      *(uint4*)buf       = v0;
      *(uint4*)(buf + 8) = v1;
#pragma unroll
      for (int j = 0; j < 16; j++) Vt[comp][(hc + j) * 32 + key] = buf[j];
    }
    __syncthreads();

    // scores: sr = Qr Kr^T + Qi Ki^T ; si = Qi Kr^T - Qr Ki^T
    v8f sr[2], si[2];
#pragma unroll
    for (int n = 0; n < 2; n++) { sr[n] = v8zero(); si[n] = v8zero(); }
#pragma unroll
    for (int c = 0; c < 2; c++) {
#pragma unroll
      for (int nt = 0; nt < 2; nt++) {
        Frag kr, ki;
        const unsigned short* pk = &Ks[0][(nt * 16 + lm) * 64 + c * 32 + lh * 8];
        kr.q[0] = *(const uint4*)pk;
        kr.q[1] = *(const uint4*)(pk + 16);
        const unsigned short* pki = &Ks[1][(nt * 16 + lm) * 64 + c * 32 + lh * 8];
        ki.q[0] = *(const uint4*)pki;
        ki.q[1] = *(const uint4*)(pki + 16);
        sr[nt] = wmma_bf16(qr[c].v, kr.v, sr[nt]);
        sr[nt] = wmma_bf16(qi[c].v, ki.v, sr[nt]);
        si[nt] = wmma_bf16(qi[c].v, kr.v, si[nt]);
        si[nt] = wmma_bf16(qrn[c].v, ki.v, si[nt]);
      }
    }

    // |score|, causal mask, online softmax over key dimension
#pragma unroll
    for (int v = 0; v < 8; v++) {
      int qrow = q0 + v + 8 * lh;             // this lane-half's query index
      int kcol = kt * 32 + lm;
      float m0 = sqrtf(sr[0][v] * sr[0][v] + si[0][v] * si[0][v]);
      float m1 = sqrtf(sr[1][v] * sr[1][v] + si[1][v] * si[1][v]);
      float p0 = (kcol <= qrow) ? m0 : -1e9f;
      float p1 = (kcol + 16 <= qrow) ? m1 : -1e9f;
      float rm = fmaxf(p0, p1);
#pragma unroll
      for (int s = 1; s < 16; s <<= 1) rm = fmaxf(rm, __shfl_xor(rm, s, 32));
      float nm = fmaxf(mrun[v], rm);
      float sc = __expf(mrun[v] - nm);
      float e0 = __expf(p0 - nm);
      float e1 = __expf(p1 - nm);
      float rs = e0 + e1;
#pragma unroll
      for (int s = 1; s < 16; s <<= 1) rs += __shfl_xor(rs, s, 32);
      lrun[v] = lrun[v] * sc + rs;
      mrun[v] = nm;
#pragma unroll
      for (int c = 0; c < 2; c++)
#pragma unroll
        for (int nt = 0; nt < 4; nt++) oacc[c][nt][v] *= sc;
      int r = v + 8 * lh;                     // relayout P through LDS
      Ps[w][r * 32 + lm]      = f2b(e0);
      Ps[w][r * 32 + 16 + lm] = f2b(e1);
    }

    // P (16x32 keys) as A-frag, V as B-frags: O += P·V
    Frag pf;
    {
      const unsigned short* pp = &Ps[w][lm * 32 + lh * 8];
      pf.q[0] = *(const uint4*)pp;
      pf.q[1] = *(const uint4*)(pp + 16);
    }
#pragma unroll
    for (int c = 0; c < 2; c++) {
#pragma unroll
      for (int nt = 0; nt < 4; nt++) {
        Frag vf;
        const unsigned short* pv = &Vt[c][(nt * 16 + lm) * 32 + lh * 8];
        vf.q[0] = *(const uint4*)pv;
        vf.q[1] = *(const uint4*)(pv + 16);
        oacc[c][nt] = wmma_bf16(pf.v, vf.v, oacc[c][nt]);
      }
    }
    __syncthreads();
  }

  // normalize and emit bf16 directly in [Xr|Xi] layout (feeds the Wo GEMM)
#pragma unroll
  for (int c = 0; c < 2; c++) {
#pragma unroll
    for (int nt = 0; nt < 4; nt++) {
#pragma unroll
      for (int v = 0; v < 8; v++) {
        float val = oacc[c][nt][v] / lrun[v];
        size_t row = (size_t)(b * Tseq + q0 + v + 8 * lh);
        int col = c * Dm + h * HDd + nt * 16 + lm;
        aattn[row * (2 * Dm) + col] = f2b(val);
      }
    }
  }
}

// ---------------------------------------------------------------------------
// z = x + attn_out  (residual 1)
// ---------------------------------------------------------------------------
__global__ void residual_kernel(const float* __restrict__ x,
                                const float* __restrict__ Co,
                                float* __restrict__ zr, float* __restrict__ zi) {
  int idx = blockIdx.x * 256 + threadIdx.x;  // over Mtok*Dm
  int m = idx >> 10, d = idx & 1023;
  size_t o = (size_t)m * (2 * Dm) + d;
  zr[idx] = x[idx] + Co[o];
  zi[idx] = x[(size_t)Mtok * Dm + idx] + Co[o + Dm];
}

// ---------------------------------------------------------------------------
// Phase-only gate + residual 2 + output: out = z + h * g/|g|
// ---------------------------------------------------------------------------
__global__ void gate_final_kernel(const float* __restrict__ zr,
                                  const float* __restrict__ zi,
                                  const float* __restrict__ Ch,
                                  const float* __restrict__ Cg,
                                  float* __restrict__ out) {
  int idx = blockIdx.x * 256 + threadIdx.x;  // over Mtok*Dm
  int m = idx >> 10, d = idx & 1023;
  size_t o = (size_t)m * (2 * Dm) + d;
  float hr = Ch[o], hi = Ch[o + Dm];
  float gr = Cg[o], gi = Cg[o + Dm];
  float inv = 1.f / (sqrtf(gr * gr + gi * gi) + 1e-8f);
  float pr = gr * inv, pi = gi * inv;
  out[idx]                       = zr[idx] + (hr * pr - hi * pi);
  out[(size_t)Mtok * Dm + idx]   = zi[idx] + (hr * pi + hi * pr);
}

// ---------------------------------------------------------------------------
// Host-side orchestration
// ---------------------------------------------------------------------------
extern "C" void kernel_launch(void* const* d_in, const int* in_sizes, int n_in,
                              void* d_out, int out_size, void* d_ws, size_t ws_size,
                              hipStream_t stream) {
  const float* x   = (const float*)d_in[0];
  const float* wq  = (const float*)d_in[2];  const float* bq  = (const float*)d_in[3];
  const float* wk  = (const float*)d_in[4];  const float* bk  = (const float*)d_in[5];
  const float* wv  = (const float*)d_in[6];  const float* bv  = (const float*)d_in[7];
  const float* wo  = (const float*)d_in[8];  const float* bo  = (const float*)d_in[9];
  const float* wf1 = (const float*)d_in[10]; const float* bf1 = (const float*)d_in[11];
  const float* wf2 = (const float*)d_in[12]; const float* bf2 = (const float*)d_in[13];
  const float* wg  = (const float*)d_in[14]; const float* bg  = (const float*)d_in[15];
  const float* g1  = (const float*)d_in[16]; const float* b1  = (const float*)d_in[17];
  const float* g2  = (const float*)d_in[18]; const float* b2  = (const float*)d_in[19];
  (void)in_sizes; (void)n_in; (void)out_size; (void)ws_size;

  char* ws = (char*)d_ws;
  const size_t MiB = 1ull << 20;
  // Packed weights (persistent across the call): 104 MiB
  unsigned short* W_Q  = (unsigned short*)(ws + 0 * MiB);
  unsigned short* W_K  = (unsigned short*)(ws + 8 * MiB);
  unsigned short* W_V  = (unsigned short*)(ws + 16 * MiB);
  unsigned short* W_O  = (unsigned short*)(ws + 24 * MiB);
  unsigned short* W_G  = (unsigned short*)(ws + 32 * MiB);
  unsigned short* W_F1 = (unsigned short*)(ws + 40 * MiB);   // 32 MiB
  unsigned short* W_F2 = (unsigned short*)(ws + 72 * MiB);   // 32 MiB
  // Activation arena with liveness-based reuse (peak ~296 MiB total)
  unsigned short* A1    = (unsigned short*)(ws + 104 * MiB); // 16 MiB
  float*          CQ    = (float*)(ws + 120 * MiB);          // 32 MiB
  float*          CK    = (float*)(ws + 152 * MiB);          // 32 MiB
  float*          CV    = (float*)(ws + 184 * MiB);          // 32 MiB
  unsigned short* QB    = (unsigned short*)(ws + 216 * MiB); // 16 MiB
  unsigned short* KB    = (unsigned short*)(ws + 232 * MiB); // 16 MiB
  unsigned short* VB    = (unsigned short*)(ws + 248 * MiB); // 16 MiB
  unsigned short* AATTN = (unsigned short*)(ws + 264 * MiB); // 16 MiB
  float*          CO    = (float*)(ws + 120 * MiB);          // reuse CQ
  float*          ZR    = (float*)(ws + 152 * MiB);          // reuse CK
  float*          ZI    = (float*)(ws + 168 * MiB);
  unsigned short* A2    = (unsigned short*)(ws + 104 * MiB); // reuse A1
  unsigned short* A3    = (unsigned short*)(ws + 184 * MiB); // 64 MiB, reuse CV+QB+KB
  float*          CH2   = (float*)(ws + 120 * MiB);          // reuse CO
  unsigned short* A4    = (unsigned short*)(ws + 248 * MiB); // reuse VB
  float*          CG    = (float*)(ws + 264 * MiB);          // 32 MiB, reuse AATTN+

  dim3 blk(256);
  auto packW = [&](const float* src, unsigned short* dst, int O, int K) {
    size_t total = 4ull * O * K;
    pack_w_kernel<<<dim3((unsigned)((total + 255) / 256)), blk, 0, stream>>>(src, dst, O, K);
  };
  auto gemm = [&](const unsigned short* A_, const unsigned short* W_,
                  const float* bias_, float* Cf_, unsigned short* Cb_,
                  int Nd, int Kd, int mode) {
    gemm_cplx_bf16<<<dim3(Nd / GBN, Mtok / GBM), blk, 0, stream>>>(
        A_, W_, bias_, Cf_, Cb_, Nd, Kd, mode);
  };

  // 0) pack all complex weights into real bf16 block form
  packW(wq, W_Q, Dm, Dm);   packW(wk, W_K, Dm, Dm);
  packW(wv, W_V, Dm, Dm);   packW(wo, W_O, Dm, Dm);
  packW(wg, W_G, Dm, Dm);
  packW(wf1, W_F1, DFFd, Dm);
  packW(wf2, W_F2, Dm, DFFd);

  const unsigned EW = (unsigned)((size_t)Mtok * Dm / 256);  // elementwise grid

  // 1) pre-norm -> A1
  cln_kernel<<<Mtok, blk, 0, stream>>>(x, x + (size_t)Mtok * Dm, g1, b1, A1);
  // 2) q,k,v projections (complex bias packed == input [2,O] layout)
  gemm(A1, W_Q, bq, CQ, nullptr, 2 * Dm, 2 * Dm, 0);
  gemm(A1, W_K, bk, CK, nullptr, 2 * Dm, 2 * Dm, 0);
  gemm(A1, W_V, bv, CV, nullptr, 2 * Dm, 2 * Dm, 0);
  // 3) RoPE + scale + bf16
  rope_convert_kernel<<<EW, blk, 0, stream>>>(CQ, CK, CV, QB, KB, VB);
  // 4) flash attention -> bf16 [Xr|Xi]
  attn_kernel<<<dim3(Tseq / 128, Hh, Bsz), blk, 0, stream>>>(QB, KB, VB, AATTN);
  // 5) output projection + residual
  gemm(AATTN, W_O, bo, CO, nullptr, 2 * Dm, 2 * Dm, 0);
  residual_kernel<<<EW, blk, 0, stream>>>(x, CO, ZR, ZI);
  // 6) FFN: norm -> f1 (fused CReLU -> bf16) -> f2 (f32 + bf16) -> gate proj
  cln_kernel<<<Mtok, blk, 0, stream>>>(ZR, ZI, g2, b2, A2);
  gemm(A2, W_F1, bf1, nullptr, A3, 2 * DFFd, 2 * Dm, 1);
  gemm(A3, W_F2, bf2, CH2, A4, 2 * Dm, 2 * DFFd, 2);
  gemm(A4, W_G,  bg,  CG, nullptr, 2 * Dm, 2 * Dm, 0);
  // 7) phase-only gate + residual -> output [2,B,T,D]
  gate_final_kernel<<<EW, blk, 0, stream>>>(ZR, ZI, CH2, CG, (float*)d_out);
}
